// RoPEMultiheadAttention_62002147885086
// MI455X (gfx1250) — matrix operand order repaired and checked by reading, not compiled, not run on hardware
//
#include <hip/hip_runtime.h>
#include <hip/hip_bf16.h>

typedef _Float16 v8h  __attribute__((ext_vector_type(8)));
typedef _Float16 v16h __attribute__((ext_vector_type(16)));
typedef float    v8f  __attribute__((ext_vector_type(8)));

#define EMBED 2048
#define HEADS 16
#define HDIM  128
#define SEQ   2048
#define BATCH 2

// ---------------------------------------------------------------------------
// f32 -> f16 convert, 8 elems/thread (b128 in, b128 out)
// ---------------------------------------------------------------------------
__global__ __launch_bounds__(256) void to_f16_kernel(const float* __restrict__ src,
                                                     _Float16* __restrict__ dst, int n) {
  int idx = (blockIdx.x * 256 + threadIdx.x) * 8;
  if (idx >= n) return;
  float4 a = *(const float4*)(src + idx);
  float4 b = *(const float4*)(src + idx + 4);
  v8h o;
  o[0] = (_Float16)a.x; o[1] = (_Float16)a.y; o[2] = (_Float16)a.z; o[3] = (_Float16)a.w;
  o[4] = (_Float16)b.x; o[5] = (_Float16)b.y; o[6] = (_Float16)b.z; o[7] = (_Float16)b.w;
  *(v8h*)(dst + idx) = o;
}

// ---------------------------------------------------------------------------
// WMMA GEMM: C[M,N] = A[M,K] * B[N,K]^T + bias[N]   (f16 in, f32 out)
// block 256 thr = 8 waves; block tile 64x256; wave tile 32x64; K step 32.
// Per wave per K-step: 8 WMMA vs 12 ds_load_b128.
// ---------------------------------------------------------------------------
__global__ __launch_bounds__(256) void gemm_f16_kernel(const _Float16* __restrict__ A,
                                                       const _Float16* __restrict__ Bw,
                                                       const float* __restrict__ bias,
                                                       float* __restrict__ C,
                                                       int M, int N, int K) {
  __shared__ __align__(16) _Float16 As[64][40];    // 80B row stride: 16B aligned
  __shared__ __align__(16) _Float16 Bs[256][40];

  const int tid  = threadIdx.x;
  const int lane = tid & 31;
  const int widx = tid >> 5;
  const int wm   = widx >> 2;      // 0..1  (M)
  const int wn   = widx & 3;       // 0..3  (N)
  const int m0   = blockIdx.y * 64;
  const int n0   = blockIdx.x * 256;
  const int ml   = lane & 15;
  const int hf   = lane >> 4;

  v8f acc[2][4] = {};

  for (int kk = 0; kk < K; kk += 32) {
    // stage A tile 64x32 (16B per thread)
    {
      int row = tid >> 2, seg = tid & 3;
      v8h v = *(const v8h*)(A + (size_t)(m0 + row) * K + kk + seg * 8);
      *(v8h*)&As[row][seg * 8] = v;
    }
    // stage B tile 256x32 (64B contiguous per thread = its own row)
    {
      const v8h* src = (const v8h*)(Bw + (size_t)(n0 + tid) * K + kk);
      *(v8h*)&Bs[tid][0]  = src[0];
      *(v8h*)&Bs[tid][8]  = src[1];
      *(v8h*)&Bs[tid][16] = src[2];
      *(v8h*)&Bs[tid][24] = src[3];
    }
    __syncthreads();

    v16h a[2], b[4];
#pragma unroll
    for (int t = 0; t < 2; ++t) {      // A frag: lane ml = row, chunks hf*8 / +16
      int r = wm * 32 + t * 16 + ml;
      v8h lo = *(const v8h*)&As[r][hf * 8];
      v8h hi = *(const v8h*)&As[r][hf * 8 + 16];
#pragma unroll
      for (int i = 0; i < 8; ++i) { a[t][i] = lo[i]; a[t][i + 8] = hi[i]; }
    }
#pragma unroll
    for (int t = 0; t < 4; ++t) {      // B frag: lane ml = col(N), chunks hf*8 / +16
      int r = wn * 64 + t * 16 + ml;
      v8h lo = *(const v8h*)&Bs[r][hf * 8];
      v8h hi = *(const v8h*)&Bs[r][hf * 8 + 16];
#pragma unroll
      for (int i = 0; i < 8; ++i) { b[t][i] = lo[i]; b[t][i + 8] = hi[i]; }
    }
#pragma unroll
    for (int tm = 0; tm < 2; ++tm)
#pragma unroll
      for (int tn = 0; tn < 4; ++tn)
        acc[tm][tn] = __builtin_amdgcn_wmma_f32_16x16x32_f16(
            false, a[tm], false, b[tn], (short)0, acc[tm][tn], false, false);
    __syncthreads();
  }

  // epilogue: bias + store f32
#pragma unroll
  for (int tm = 0; tm < 2; ++tm)
#pragma unroll
    for (int tn = 0; tn < 4; ++tn) {
      int col = n0 + wn * 64 + tn * 16 + ml;
      float bn = bias[col];
#pragma unroll
      for (int r = 0; r < 8; ++r) {
        int row = m0 + wm * 32 + tm * 16 + r + hf * 8;
        C[(size_t)row * N + col] = acc[tm][tn][r] + bn;
      }
    }
}

// ---------------------------------------------------------------------------
// RoPE + repack: src f32 (B,S,E).  mode 0: rope -> (B,H,S,D) f16
//                                  mode 1: plain -> (B,H,D,S) f16 (V transposed)
// ---------------------------------------------------------------------------
__global__ __launch_bounds__(256) void rope_pack_kernel(const float* __restrict__ src,
                                                        _Float16* __restrict__ dst, int mode) {
  int idx = blockIdx.x * 256 + threadIdx.x;          // over B*S*E
  int e = idx & (EMBED - 1);
  int t = idx >> 11;                                  // b*S + s
  int s = t & (SEQ - 1);
  int b = t >> 11;
  int h = e >> 7;
  int d = e & (HDIM - 1);
  float v = src[idx];
  if (mode == 0) {
    int i = d & 63;
    float inv = __expf(-0.14391157f * (float)i);      // 10000^(-i/64)
    float ang = (float)s * inv;
    float sn, cs;
    __sincosf(ang, &sn, &cs);
    float rot = (d < 64) ? -src[idx + 64] : src[idx - 64];
    v = v * cs + rot * sn;
    dst[((size_t)(b * HEADS + h) * SEQ + s) * HDIM + d] = (_Float16)v;
  } else {
    dst[((size_t)(b * HEADS + h) * HDIM + d) * SEQ + s] = (_Float16)v;
  }
}

// ---------------------------------------------------------------------------
// Flash attention: one wave = one 32-row Q tile of one (b,h).
// K/V fragments are reused across both 16-row M-tiles: per 32-key strip
// 32 WMMAs vs 32 global b128 loads.
// Qh,Kh: (B,H,S,D) f16.  Vt: (B,H,D,S) f16.  Oh: (B,S,E) f16.
// ---------------------------------------------------------------------------
__global__ __launch_bounds__(128) void flash_kernel(const _Float16* __restrict__ Qh,
                                                    const _Float16* __restrict__ Kh,
                                                    const _Float16* __restrict__ Vt,
                                                    _Float16* __restrict__ Oh) {
  const int lane = threadIdx.x & 31;
  const int wv   = threadIdx.x >> 5;
  const int gw   = blockIdx.x * 4 + wv;               // 0..2047
  const int qt   = gw & 63;                           // 64 q tiles (32 rows) per (b,h)
  const int bh   = gw >> 6;                           // 0..31
  const int b    = bh >> 4;
  const int h    = bh & 15;
  const int ml   = lane & 15;
  const int hf   = lane >> 4;

  const _Float16* Qb = Qh + (size_t)bh * SEQ * HDIM;
  const _Float16* Kb = Kh + (size_t)bh * SEQ * HDIM;
  const _Float16* Vb = Vt + (size_t)bh * HDIM * SEQ;

  __shared__ __align__(16) _Float16 Pl[4][32][40];    // wave-private P strips

  // load Q A-fragments (32x128), pre-scaled by D^-1/2
  const float scale = 0.08838834764831845f;
  const int sq = qt * 32;
  v16h qa[2][4];
#pragma unroll
  for (int mt = 0; mt < 2; ++mt)
#pragma unroll
    for (int c = 0; c < 4; ++c) {
      const _Float16* p = Qb + (size_t)(sq + mt * 16 + ml) * HDIM + c * 32 + hf * 8;
      v8h lo = *(const v8h*)p;
      v8h hi = *(const v8h*)(p + 16);
#pragma unroll
      for (int i = 0; i < 8; ++i) {
        qa[mt][c][i]     = (_Float16)((float)lo[i] * scale);
        qa[mt][c][i + 8] = (_Float16)((float)hi[i] * scale);
      }
    }

  v8f o[2][8] = {};
  float mr[2][8], lr[2][8];
#pragma unroll
  for (int mt = 0; mt < 2; ++mt)
#pragma unroll
    for (int r = 0; r < 8; ++r) { mr[mt][r] = -__builtin_inff(); lr[mt][r] = 0.f; }

  for (int j = 0; j < SEQ; j += 32) {
    v8f s[2][2] = {};
#pragma unroll
    for (int c = 0; c < 4; ++c) {
      const _Float16* p0 = Kb + (size_t)(j + ml) * HDIM + c * 32 + hf * 8;
      v8h l0 = *(const v8h*)p0;
      v8h h0 = *(const v8h*)(p0 + 16);
      v16h b0;
#pragma unroll
      for (int i = 0; i < 8; ++i) { b0[i] = l0[i]; b0[i + 8] = h0[i]; }

      const _Float16* p1 = Kb + (size_t)(j + 16 + ml) * HDIM + c * 32 + hf * 8;
      v8h l1 = *(const v8h*)p1;
      v8h h1 = *(const v8h*)(p1 + 16);
      v16h b1;
#pragma unroll
      for (int i = 0; i < 8; ++i) { b1[i] = l1[i]; b1[i + 8] = h1[i]; }

#pragma unroll
      for (int mt = 0; mt < 2; ++mt) {
        s[mt][0] = __builtin_amdgcn_wmma_f32_16x16x32_f16(false, qa[mt][c], false, b0,
                                                          (short)0, s[mt][0], false, false);
        s[mt][1] = __builtin_amdgcn_wmma_f32_16x16x32_f16(false, qa[mt][c], false, b1,
                                                          (short)0, s[mt][1], false, false);
      }
    }

    // streaming softmax over the 32-wide strip; reductions within 16-lane halves
    float alpha[2][8];
#pragma unroll
    for (int mt = 0; mt < 2; ++mt)
#pragma unroll
      for (int r = 0; r < 8; ++r) {
        float mx = fmaxf(s[mt][0][r], s[mt][1][r]);
#pragma unroll
        for (int off = 1; off < 16; off <<= 1) mx = fmaxf(mx, __shfl_xor(mx, off, 32));
        float mn = fmaxf(mr[mt][r], mx);
        float al = __expf(mr[mt][r] - mn);
        float p0 = __expf(s[mt][0][r] - mn);
        float p1 = __expf(s[mt][1][r] - mn);
        float rs = p0 + p1;
#pragma unroll
        for (int off = 1; off < 16; off <<= 1) rs += __shfl_xor(rs, off, 32);
        lr[mt][r] = lr[mt][r] * al + rs;
        mr[mt][r] = mn;
        alpha[mt][r] = al;
        Pl[wv][mt * 16 + r + hf * 8][ml]      = (_Float16)p0;
        Pl[wv][mt * 16 + r + hf * 8][16 + ml] = (_Float16)p1;
      }

    // re-read P as two 16x32 A-fragments (per-wave LDS ops complete in order)
    v16h pa[2];
#pragma unroll
    for (int mt = 0; mt < 2; ++mt) {
      const _Float16* p = &Pl[wv][mt * 16 + ml][hf * 8];
      v8h lo = *(const v8h*)p;
      v8h hi = *(const v8h*)(p + 16);
#pragma unroll
      for (int i = 0; i < 8; ++i) { pa[mt][i] = lo[i]; pa[mt][i + 8] = hi[i]; }
    }

    // rescale accumulators, then P@V over 8 d-chunks (V frag shared across M-tiles)
#pragma unroll
    for (int mt = 0; mt < 2; ++mt)
#pragma unroll
      for (int t = 0; t < 8; ++t)
#pragma unroll
        for (int r = 0; r < 8; ++r) o[mt][t][r] *= alpha[mt][r];

#pragma unroll
    for (int t = 0; t < 8; ++t) {
      const _Float16* p = Vb + (size_t)(t * 16 + ml) * SEQ + j + hf * 8;
      v8h lo = *(const v8h*)p;
      v8h hi = *(const v8h*)(p + 16);
      v16h vb;
#pragma unroll
      for (int i = 0; i < 8; ++i) { vb[i] = lo[i]; vb[i + 8] = hi[i]; }
#pragma unroll
      for (int mt = 0; mt < 2; ++mt)
        o[mt][t] = __builtin_amdgcn_wmma_f32_16x16x32_f16(false, pa[mt], false, vb,
                                                          (short)0, o[mt][t], false, false);
    }
  }

  // finalize: divide by l, store f16 into (B,S,E) layout for the O-projection GEMM
#pragma unroll
  for (int mt = 0; mt < 2; ++mt)
#pragma unroll
    for (int t = 0; t < 8; ++t)
#pragma unroll
      for (int r = 0; r < 8; ++r) {
        float v = o[mt][t][r] / lr[mt][r];
        int row = sq + mt * 16 + r + hf * 8;
        Oh[((size_t)(b * SEQ + row)) * EMBED + h * HDIM + t * 16 + ml] = (_Float16)v;
      }
}

// ---------------------------------------------------------------------------
extern "C" void kernel_launch(void* const* d_in, const int* in_sizes, int n_in,
                              void* d_out, int out_size, void* d_ws, size_t ws_size,
                              hipStream_t stream) {
  (void)in_sizes; (void)n_in; (void)out_size; (void)ws_size;
  const float* x  = (const float*)d_in[0];
  const float* Wq = (const float*)d_in[1];
  const float* bq = (const float*)d_in[2];
  const float* Wk = (const float*)d_in[3];
  const float* bk = (const float*)d_in[4];
  const float* Wv = (const float*)d_in[5];
  const float* bv = (const float*)d_in[6];
  const float* Wo = (const float*)d_in[7];
  const float* bo = (const float*)d_in[8];

  const int M = BATCH * SEQ;                // 4096
  const size_t nX = (size_t)M * EMBED;      // 8,388,608
  const size_t nW = (size_t)EMBED * EMBED;  // 4,194,304

  char* wp = (char*)d_ws;
  _Float16* xh  = (_Float16*)wp; wp += nX * sizeof(_Float16);
  _Float16* Wqh = (_Float16*)wp; wp += nW * sizeof(_Float16);
  _Float16* Wkh = (_Float16*)wp; wp += nW * sizeof(_Float16);
  _Float16* Wvh = (_Float16*)wp; wp += nW * sizeof(_Float16);
  _Float16* Woh = (_Float16*)wp; wp += nW * sizeof(_Float16);
  float*    tmp = (float*)wp;    wp += nX * sizeof(float);
  _Float16* Qh  = (_Float16*)wp; wp += nX * sizeof(_Float16);
  _Float16* Kh  = (_Float16*)wp; wp += nX * sizeof(_Float16);
  _Float16* Vt  = (_Float16*)wp; wp += nX * sizeof(_Float16);
  _Float16* Oh  = (_Float16*)wp; wp += nX * sizeof(_Float16);

  // f32 -> f16 converts
  to_f16_kernel<<<(int)(nX / 2048), 256, 0, stream>>>(x,  xh,  (int)nX);
  to_f16_kernel<<<(int)(nW / 2048), 256, 0, stream>>>(Wq, Wqh, (int)nW);
  to_f16_kernel<<<(int)(nW / 2048), 256, 0, stream>>>(Wk, Wkh, (int)nW);
  to_f16_kernel<<<(int)(nW / 2048), 256, 0, stream>>>(Wv, Wvh, (int)nW);
  to_f16_kernel<<<(int)(nW / 2048), 256, 0, stream>>>(Wo, Woh, (int)nW);

  dim3 gg(EMBED / 256, M / 64);   // (8, 64)
  const int packBlocks = (int)(nX / 256);

  // Q = x Wq^T + bq ; rope ; pack (B,H,S,D)
  gemm_f16_kernel<<<gg, 256, 0, stream>>>(xh, Wqh, bq, tmp, M, EMBED, EMBED);
  rope_pack_kernel<<<packBlocks, 256, 0, stream>>>(tmp, Qh, 0);
  // K
  gemm_f16_kernel<<<gg, 256, 0, stream>>>(xh, Wkh, bk, tmp, M, EMBED, EMBED);
  rope_pack_kernel<<<packBlocks, 256, 0, stream>>>(tmp, Kh, 0);
  // V -> transposed (B,H,D,S)
  gemm_f16_kernel<<<gg, 256, 0, stream>>>(xh, Wvh, bv, tmp, M, EMBED, EMBED);
  rope_pack_kernel<<<packBlocks, 256, 0, stream>>>(tmp, Vt, 1);

  // attention: 2048 waves, 4 per block
  flash_kernel<<<512, 128, 0, stream>>>(Qh, Kh, Vt, Oh);

  // output projection straight to d_out
  gemm_f16_kernel<<<gg, 256, 0, stream>>>(Oh, Woh, bo, (float*)d_out, M, EMBED, EMBED);
}
